// MPNConditionalDecoder_42606075576982
// MI455X (gfx1250) — matrix operand order repaired and checked by reading, not compiled, not run on hardware
//
#include <hip/hip_runtime.h>
#include <hip/hip_bf16.h>

// ---------------- problem constants (from reference) ----------------
#define T_STEPS 47          // T-1 scan steps
#define B_   64
#define S_   48
#define CTX_ 1024
#define H_   1024
#define E_   512
#define V_   32000
#define H3_  (3 * H_)

typedef __bf16 bf16;
typedef bf16  v16bf __attribute__((ext_vector_type(16)));
typedef float v8f   __attribute__((ext_vector_type(8)));

union ABfrag { v16bf v; uint4 u[2]; };

__device__ __forceinline__ float fast_tanh(float x) {
  // 1 - 2/(e^{2x}+1); saturates correctly at +/-1
  return 1.f - 2.f / (__expf(2.f * x) + 1.f);
}
__device__ __forceinline__ float fast_sigmoid(float x) {
  return 1.f / (1.f + __expf(-x));
}
__device__ __forceinline__ bf16 f2bf(float x) { return (bf16)x; }

// =====================================================================
// WMMA bf16 GEMM:  C(M,N) = act( A(M,K) * W(K,N) + bias )
//   A  : bf16 row-major (M x K)
//   Wt : bf16, W transposed, row-major (N x K)  -> contiguous K per column
// One wave computes a 16x64 tile (4 x v_wmma_f32_16x16x32_bf16 per K-step),
// reusing the A fragment across the 4 N-subtiles.
// A fragment (ISA 7.12.2, 16-bit A 16x32): lane m=lane&15, half L=lane>>4
//   holds K = L*16 .. L*16+15 (contiguous order, not the ISA table; matches B)  -> two b128 loads.
// B fragment (16-bit B 32x16): lane n=lane&15, holds K = L*16 .. L*16+15
//   -> two contiguous b128 loads from Wt row n.
// =====================================================================
__global__ __launch_bounds__(256) void k_gemm_bf16(
    const bf16* __restrict__ A, const bf16* __restrict__ Wt,
    const float* __restrict__ bias,
    float* __restrict__ Cf, bf16* __restrict__ Cb,
    int M, int N, int K, int act)
{
  const int lane = threadIdx.x & 31;
  const int wid  = (int)((blockIdx.x * blockDim.x + threadIdx.x) >> 5);
  const int mTiles  = M >> 4;
  const int nGroups = N >> 6;
  if (wid >= mTiles * nGroups) return;      // wave-uniform: EXEC stays all-1s
  const int mT = wid % mTiles;
  const int nG = wid / mTiles;
  const int L  = lane >> 4;
  const int nl = lane & 15;

  const bf16* Arow = A  + (size_t)(mT * 16 + nl) * K;
  const bf16* B0   = Wt + (size_t)(nG * 64 + nl) * K + L * 16;
  const bf16* B1   = B0 + (size_t)16 * K;
  const bf16* B2   = B0 + (size_t)32 * K;
  const bf16* B3   = B0 + (size_t)48 * K;

  v8f c0 = {}, c1 = {}, c2 = {}, c3 = {};

  for (int k0 = 0; k0 < K; k0 += 32) {
    ABfrag a, b0, b1, b2, b3;
    a.u[0]  = *(const uint4*)(Arow + k0 + L * 16);
    a.u[1]  = *(const uint4*)(Arow + k0 + L * 16 + 8);
    b0.u[0] = *(const uint4*)(B0 + k0);
    b0.u[1] = *(const uint4*)(B0 + k0 + 8);
    b1.u[0] = *(const uint4*)(B1 + k0);
    b1.u[1] = *(const uint4*)(B1 + k0 + 8);
    b2.u[0] = *(const uint4*)(B2 + k0);
    b2.u[1] = *(const uint4*)(B2 + k0 + 8);
    b3.u[0] = *(const uint4*)(B3 + k0);
    b3.u[1] = *(const uint4*)(B3 + k0 + 8);

    // prefetch the streaming (weight) side ~256B ahead
    __builtin_prefetch(B0 + k0 + 128, 0, 1);
    __builtin_prefetch(B1 + k0 + 128, 0, 1);
    __builtin_prefetch(B2 + k0 + 128, 0, 1);
    __builtin_prefetch(B3 + k0 + 128, 0, 1);

    c0 = __builtin_amdgcn_wmma_f32_16x16x32_bf16(false, a.v, false, b0.v, (short)0, c0, false, false);
    c1 = __builtin_amdgcn_wmma_f32_16x16x32_bf16(false, a.v, false, b1.v, (short)0, c1, false, false);
    c2 = __builtin_amdgcn_wmma_f32_16x16x32_bf16(false, a.v, false, b2.v, (short)0, c2, false, false);
    c3 = __builtin_amdgcn_wmma_f32_16x16x32_bf16(false, a.v, false, b3.v, (short)0, c3, false, false);
  }

  // C layout (ISA): element r -> row = mT*16 + L*8 + r, col = nG*64 + subtile*16 + nl
  const int rowBase = mT * 16 + L * 8;
  const int colBase = nG * 64 + nl;
  auto emit = [&](float v, int row, int col) {
    if (bias) v += bias[col];
    if (act)  v = fast_tanh(v);
    size_t idx = (size_t)row * N + col;
    if (Cf) Cf[idx] = v;
    if (Cb) Cb[idx] = f2bf(v);
  };
  #pragma unroll
  for (int r = 0; r < 8; ++r) {
    const int row = rowBase + r;
    emit(c0[r], row, colBase +  0);
    emit(c1[r], row, colBase + 16);
    emit(c2[r], row, colBase + 32);
    emit(c3[r], row, colBase + 48);
  }
}

// ---------------- one-time conversion kernels ----------------
// W (K,N) f32  ->  Wt (N,K) bf16
__global__ __launch_bounds__(256) void k_transpose_bf16(
    const float* __restrict__ W, bf16* __restrict__ Wt, int K, int N)
{
  long long i = (long long)blockIdx.x * 256 + threadIdx.x;
  if (i >= (long long)K * N) return;
  int n = (int)(i / K);
  int k = (int)(i % K);
  Wt[i] = f2bf(W[(size_t)k * N + n]);
}

// y_emb[t,b,:] = bf16(emb_W[y[t,b]]) masked by (y!=0); t in [0, T_STEPS)
__global__ __launch_bounds__(256) void k_embed(
    const int* __restrict__ y, const float* __restrict__ embW, bf16* __restrict__ out)
{
  int i  = blockIdx.x * 256 + threadIdx.x;   // T_STEPS*B*E total, exact multiple
  int e  = i & (E_ - 1);
  int tb = i >> 9;                           // / E_
  int tok = y[tb];
  out[i] = f2bf(tok != 0 ? embW[(size_t)tok * E_ + e] : 0.f);
}

__global__ __launch_bounds__(256) void k_cvt_bf16(
    const float* __restrict__ in, bf16* __restrict__ out, int n)
{
  int i = blockIdx.x * 256 + threadIdx.x;
  if (i < n) out[i] = f2bf(in[i]);
}

// ctx_mean[b,c] = mean_s ctx[s,b,c]  (bf16 out)
__global__ __launch_bounds__(256) void k_ctx_mean(
    const float* __restrict__ ctx, bf16* __restrict__ out)
{
  int i = blockIdx.x * 256 + threadIdx.x;    // B*CTX total
  int b = i >> 10;
  int c = i & (CTX_ - 1);
  float s = 0.f;
  for (int ss = 0; ss < S_; ++ss) s += ctx[((size_t)ss * B_ + b) * CTX_ + c];
  out[i] = f2bf(s * (1.f / S_));
}

// ---------------- per-step elementwise kernels ----------------
// GRU gate combine: h' = (1-z)*n + z*h ; also emits bf16 copy of h'
__global__ __launch_bounds__(256) void k_gru(
    const float* __restrict__ gi, const float* __restrict__ gh,
    const float* __restrict__ h, float* __restrict__ ho, bf16* __restrict__ hob)
{
  int i = blockIdx.x * 256 + threadIdx.x;    // B*H total
  int b = i >> 10;
  int c = i & (H_ - 1);
  size_t g = (size_t)b * H3_ + c;
  float r = fast_sigmoid(gi[g] + gh[g]);
  float z = fast_sigmoid(gi[g + H_] + gh[g + H_]);
  float n = fast_tanh(gi[g + 2 * H_] + r * gh[g + 2 * H_]);
  float hn = (1.f - z) * n + z * h[i];
  ho[i]  = hn;
  hob[i] = f2bf(hn);
}

// scores[s*B+b] = sum_c tanh(ctx_proj[s,b,c] + hq[b,c]) * mlp[c]
__global__ __launch_bounds__(256) void k_scores(
    const float* __restrict__ ctx_proj, const float* __restrict__ hq,
    const float* __restrict__ mlp, float* __restrict__ scores)
{
  __shared__ float red[256];
  const int sb = blockIdx.x;                 // s*B + b
  const int b  = sb & (B_ - 1);
  const float* cp = ctx_proj + (size_t)sb * CTX_;
  const float* hv = hq + (size_t)b * CTX_;
  float acc = 0.f;
  for (int c = threadIdx.x; c < CTX_; c += 256)
    acc += fast_tanh(cp[c] + hv[c]) * mlp[c];
  red[threadIdx.x] = acc;
  __syncthreads();
  for (int off = 128; off > 0; off >>= 1) {
    if (threadIdx.x < off) red[threadIdx.x] += red[threadIdx.x + off];
    __syncthreads();
  }
  if (threadIdx.x == 0) scores[sb] = red[0];
}

// softmax over s, then z[b,c] = sum_s alpha[s]*ctx[s,b,c]  (bf16 out)
__global__ __launch_bounds__(256) void k_softmax_z(
    const float* __restrict__ scores, const float* __restrict__ ctx,
    bf16* __restrict__ zb)
{
  __shared__ float al[S_];
  const int b = blockIdx.x;
  if (threadIdx.x == 0) {
    float mx = -3.4e38f;
    for (int s = 0; s < S_; ++s) mx = fmaxf(mx, scores[s * B_ + b]);
    float sm = 0.f;
    for (int s = 0; s < S_; ++s) { float e = __expf(scores[s * B_ + b] - mx); al[s] = e; sm += e; }
    float inv = 1.f / sm;
    for (int s = 0; s < S_; ++s) al[s] *= inv;
  }
  __syncthreads();
  for (int c = threadIdx.x; c < CTX_; c += 256) {
    float acc = 0.f;
    for (int s = 0; s < S_; ++s)
      acc += al[s] * ctx[((size_t)s * B_ + b) * CTX_ + c];
    zb[(size_t)b * CTX_ + c] = f2bf(acc);
  }
}

// per-row log_softmax over V; writes logps and per-(t,b) masked NLL
__global__ __launch_bounds__(256) void k_logsoftmax_nll(
    const float* __restrict__ logits, const int* __restrict__ y, int t,
    float* __restrict__ outLogp, float* __restrict__ nll)
{
  __shared__ float red[256];
  const int b = blockIdx.x;
  const float* row = logits + (size_t)b * V_;
  float mx = -3.4e38f;
  for (int v = threadIdx.x; v < V_; v += 256) mx = fmaxf(mx, row[v]);
  red[threadIdx.x] = mx;
  __syncthreads();
  for (int off = 128; off > 0; off >>= 1) {
    if (threadIdx.x < off) red[threadIdx.x] = fmaxf(red[threadIdx.x], red[threadIdx.x + off]);
    __syncthreads();
  }
  mx = red[0];
  __syncthreads();
  float sm = 0.f;
  for (int v = threadIdx.x; v < V_; v += 256) sm += __expf(row[v] - mx);
  red[threadIdx.x] = sm;
  __syncthreads();
  for (int off = 128; off > 0; off >>= 1) {
    if (threadIdx.x < off) red[threadIdx.x] += red[threadIdx.x + off];
    __syncthreads();
  }
  const float lse = __logf(red[0]);
  for (int v = threadIdx.x; v < V_; v += 256)
    outLogp[(size_t)b * V_ + v] = row[v] - mx - lse;
  if (threadIdx.x == 0) {
    int tgt = y[(t + 1) * B_ + b];
    nll[b] = (tgt != 0) ? -(row[tgt] - mx - lse) : 0.f;
  }
}

// deterministic fixed-order loss reduction (no float atomics)
__global__ __launch_bounds__(256) void k_reduce_loss(
    const float* __restrict__ nll, int n, float* __restrict__ out)
{
  __shared__ float red[256];
  float s = 0.f;
  for (int i = threadIdx.x; i < n; i += 256) s += nll[i];
  red[threadIdx.x] = s;
  __syncthreads();
  for (int off = 128; off > 0; off >>= 1) {
    if (threadIdx.x < off) red[threadIdx.x] += red[threadIdx.x + off];
    __syncthreads();
  }
  if (threadIdx.x == 0) out[0] = red[0];
}

// =====================================================================
// Host orchestration
// =====================================================================
static inline void gemm(const bf16* A, const bf16* Wt, const float* bias,
                        float* Cf, bf16* Cb, int M, int N, int K, int act,
                        hipStream_t s)
{
  int waves  = (M / 16) * (N / 64);
  int blocks = (waves + 7) / 8;              // 8 waves / 256-thread block
  k_gemm_bf16<<<blocks, 256, 0, s>>>(A, Wt, bias, Cf, Cb, M, N, K, act);
}

extern "C" void kernel_launch(void* const* d_in, const int* in_sizes, int n_in,
                              void* d_out, int out_size, void* d_ws, size_t ws_size,
                              hipStream_t stream) {
  (void)in_sizes; (void)n_in; (void)out_size; (void)ws_size;
  const int*   y          = (const int*)  d_in[0];
  const float* ctx        = (const float*)d_in[1];
  const float* emb_W      = (const float*)d_in[2];
  const float* ctx2ctx_W  = (const float*)d_in[3];
  const float* hid2ctx_W  = (const float*)d_in[4];
  const float* mlp_W      = (const float*)d_in[5];
  const float* dec_init_W = (const float*)d_in[6];
  const float* dec_init_b = (const float*)d_in[7];
  const float* dec0_Wih   = (const float*)d_in[8];
  const float* dec0_Whh   = (const float*)d_in[9];
  const float* dec0_bih   = (const float*)d_in[10];
  const float* dec0_bhh   = (const float*)d_in[11];
  const float* dec1_Wih   = (const float*)d_in[12];
  const float* dec1_Whh   = (const float*)d_in[13];
  const float* dec1_bih   = (const float*)d_in[14];
  const float* dec1_bhh   = (const float*)d_in[15];
  const float* hid2out_W  = (const float*)d_in[16];
  const float* hid2out_b  = (const float*)d_in[17];
  const float* out2prob_W = (const float*)d_in[18];
  const float* out2prob_b = (const float*)d_in[19];

  float* out = (float*)d_out;                // [0]=loss, [1..]=logps (T-1,B,V)

  // ---- workspace carve-out (256B aligned) ----
  char* w = (char*)d_ws;
  auto alloc = [&](size_t bytes) -> char* {
    char* p = w; w += (bytes + 255) & ~(size_t)255; return p;
  };
  bf16* wt_d0_ih  = (bf16*)alloc((size_t)H3_ * E_   * 2);  // (3072,512)
  bf16* wt_d0_hh  = (bf16*)alloc((size_t)H3_ * H_   * 2);  // (3072,1024)
  bf16* wt_d1_ih  = (bf16*)alloc((size_t)H3_ * CTX_ * 2);
  bf16* wt_d1_hh  = (bf16*)alloc((size_t)H3_ * H_   * 2);
  bf16* wt_h2c    = (bf16*)alloc((size_t)CTX_ * H_  * 2);
  bf16* wt_h2o    = (bf16*)alloc((size_t)E_  * H_   * 2);  // (512,1024)
  bf16* wt_o2p    = (bf16*)alloc((size_t)V_  * E_   * 2);  // (32000,512)
  bf16* wt_c2c    = (bf16*)alloc((size_t)CTX_ * CTX_* 2);
  bf16* wt_dini   = (bf16*)alloc((size_t)H_  * CTX_ * 2);
  bf16* y_emb     = (bf16*)alloc((size_t)T_STEPS * B_ * E_ * 2);
  bf16* ctx_b16   = (bf16*)alloc((size_t)S_ * B_ * CTX_ * 2);
  float* ctx_proj = (float*)alloc((size_t)S_ * B_ * CTX_ * 4);
  bf16* ctx_mean  = (bf16*)alloc((size_t)B_ * CTX_ * 2);
  float* hA_f     = (float*)alloc((size_t)B_ * H_ * 4);
  bf16*  hA_b     = (bf16*) alloc((size_t)B_ * H_ * 2);
  float* hB_f     = (float*)alloc((size_t)B_ * H_ * 4);
  bf16*  hB_b     = (bf16*) alloc((size_t)B_ * H_ * 2);
  float* h1_f     = (float*)alloc((size_t)B_ * H_ * 4);
  bf16*  h1_b     = (bf16*) alloc((size_t)B_ * H_ * 2);
  float* gi       = (float*)alloc((size_t)B_ * H3_ * 4);
  float* gh       = (float*)alloc((size_t)B_ * H3_ * 4);
  float* hq       = (float*)alloc((size_t)B_ * CTX_ * 4);
  float* scores   = (float*)alloc((size_t)S_ * B_ * 4);
  bf16*  z_b16    = (bf16*) alloc((size_t)B_ * CTX_ * 2);
  bf16*  logit_b  = (bf16*) alloc((size_t)B_ * E_ * 2);
  float* logits_v = (float*)alloc((size_t)B_ * V_ * 4);
  float* nll      = (float*)alloc((size_t)T_STEPS * B_ * 4);

  auto tr = [&](const float* W, bf16* Wt, int K, int N) {
    long long n = (long long)K * N;
    k_transpose_bf16<<<(unsigned)((n + 255) / 256), 256, 0, stream>>>(W, Wt, K, N);
  };

  // ---- one-time weight conversion (fp32 -> bf16 transposed) ----
  tr(dec0_Wih,   wt_d0_ih, E_,   H3_);
  tr(dec0_Whh,   wt_d0_hh, H_,   H3_);
  tr(dec1_Wih,   wt_d1_ih, CTX_, H3_);
  tr(dec1_Whh,   wt_d1_hh, H_,   H3_);
  tr(hid2ctx_W,  wt_h2c,   H_,   CTX_);
  tr(hid2out_W,  wt_h2o,   H_,   E_);
  tr(out2prob_W, wt_o2p,   E_,   V_);
  tr(ctx2ctx_W,  wt_c2c,   CTX_, CTX_);
  tr(dec_init_W, wt_dini,  CTX_, H_);

  // ---- precompute activations ----
  k_embed<<<(T_STEPS * B_ * E_) / 256, 256, 0, stream>>>(y, emb_W, y_emb);
  {
    int n = S_ * B_ * CTX_;
    k_cvt_bf16<<<(n + 255) / 256, 256, 0, stream>>>(ctx, ctx_b16, n);
  }
  k_ctx_mean<<<(B_ * CTX_) / 256, 256, 0, stream>>>(ctx, ctx_mean);

  // ctx_proj = ctx @ ctx2ctx_W   (3072 x 1024 x 1024)
  gemm(ctx_b16, wt_c2c, nullptr, ctx_proj, nullptr, S_ * B_, CTX_, CTX_, 0, stream);
  // h0 = tanh(ctx_mean @ dec_init_W + b)
  gemm(ctx_mean, wt_dini, dec_init_b, hA_f, hA_b, B_, H_, CTX_, 1, stream);

  float* hcur_f = hA_f;  bf16* hcur_b = hA_b;
  float* hnxt_f = hB_f;  bf16* hnxt_b = hB_b;

  // ---- sequential scan over 47 steps ----
  for (int t = 0; t < T_STEPS; ++t) {
    // GRU0 gates
    gemm(y_emb + (size_t)t * B_ * E_, wt_d0_ih, dec0_bih, gi, nullptr, B_, H3_, E_, 0, stream);
    gemm(hcur_b,                      wt_d0_hh, dec0_bhh, gh, nullptr, B_, H3_, H_, 0, stream);
    k_gru<<<(B_ * H_) / 256, 256, 0, stream>>>(gi, gh, hcur_f, h1_f, h1_b);

    // attention
    gemm(h1_b, wt_h2c, nullptr, hq, nullptr, B_, CTX_, H_, 0, stream);
    k_scores<<<S_ * B_, 256, 0, stream>>>(ctx_proj, hq, mlp_W, scores);
    k_softmax_z<<<B_, 256, 0, stream>>>(scores, ctx, z_b16);

    // GRU1 gates
    gemm(z_b16, wt_d1_ih, dec1_bih, gi, nullptr, B_, H3_, CTX_, 0, stream);
    gemm(h1_b,  wt_d1_hh, dec1_bhh, gh, nullptr, B_, H3_, H_,   0, stream);
    k_gru<<<(B_ * H_) / 256, 256, 0, stream>>>(gi, gh, h1_f, hnxt_f, hnxt_b);

    // logit = tanh(h2 @ hid2out + b)  (bf16 only, feeds vocab GEMM)
    gemm(hnxt_b, wt_h2o, hid2out_b, nullptr, logit_b, B_, E_, H_, 1, stream);
    // vocab projection (64 x 512 x 32000)
    gemm(logit_b, wt_o2p, out2prob_b, logits_v, nullptr, B_, V_, E_, 0, stream);

    k_logsoftmax_nll<<<B_, 256, 0, stream>>>(
        logits_v, y, t, out + 1 + (size_t)t * B_ * V_, nll + (size_t)t * B_);

    // swap h ping-pong
    float* tf = hcur_f; hcur_f = hnxt_f; hnxt_f = tf;
    bf16*  tb = hcur_b; hcur_b = hnxt_b; hnxt_b = tb;
  }

  k_reduce_loss<<<1, 256, 0, stream>>>(nll, T_STEPS * B_, out);
}
